// HRoPEBlock_48988396978515
// MI455X (gfx1250) — matrix operand order repaired and checked by reading, not compiled, not run on hardware
//
#include <hip/hip_runtime.h>
#include <hip/hip_bf16.h>
#include <cstdint>
#include <cstddef>
#include <math.h>

// ---------------------------------------------------------------------------
// Types for CDNA5 WMMA (wave32, 16x16x32 bf16 -> f32) and TDM descriptors
// ---------------------------------------------------------------------------
typedef __bf16 bf16_t;
typedef __attribute__((ext_vector_type(16))) bf16_t v16bf;
typedef __attribute__((ext_vector_type(8)))  bf16_t v8bf;
typedef __attribute__((ext_vector_type(8)))  float  v8f;
typedef __attribute__((ext_vector_type(4)))  unsigned int u32x4;
typedef __attribute__((ext_vector_type(8)))  int i32x8;
typedef __attribute__((ext_vector_type(4)))  int i32x4;

#define S_LEN   2048
#define D_MODEL 2048
#define NHEAD   16
#define HDIM    128
#define FF_DIM  8192
#define RMS_EPS 1.1920929e-07f   // jnp.finfo(float32).eps
#define ATTN_EPS 1e-6f

// GEMM LDS tiling: 128 rows x 32 bf16 data (64B) + 32B TDM pad -> 96B pitch
#define TILE_PITCH 48                         // bf16 elements per LDS row
#define TILE_ELEMS (128 * TILE_PITCH)         // 6144 elems = 12288 B
#define TILE_BYTES (TILE_ELEMS * 2)
#define GEMM_LDS_BYTES (4 * TILE_BYTES)       // 2 buffers x (A + W) = 48 KB

#if defined(__has_builtin)
#if __has_builtin(__builtin_amdgcn_tensor_load_to_lds) && \
    __has_builtin(__builtin_amdgcn_s_wait_tensorcnt)
#define USE_TDM 1
#endif
#endif
#ifndef USE_TDM
#define USE_TDM 0
#endif

// ---------------------------------------------------------------------------
// A-fragment loader: 16x32 bf16 A-matrix, contiguous k layout:
//   lanes 0-15  (row M=lane):    elements 0..15 = K 0..15
//   lanes 16-31 (row M=lane-16): elements 0..15 = K 16..31
// Caller passes p = rowBase + half*16  (half = lane>>4).
// ---------------------------------------------------------------------------
__device__ inline v16bf load_a_frag(const bf16_t* p) {
  v8bf lo = *(const v8bf*)(p);
  v8bf hi = *(const v8bf*)(p + 8);
  v16bf r;
#pragma unroll
  for (int i = 0; i < 8; ++i) { r[i] = lo[i]; r[i + 8] = hi[i]; }
  return r;
}

#if USE_TDM
// ---------------------------------------------------------------------------
// Issue one TDM tensor_load_to_lds of a 128-row x 32-col bf16 tile from a
// row-major [nrows, K] matrix into LDS at byte offset lds_off. D# fields per
// ISA ch.8: group0 = {count, lds_addr, global_addr, type=2}; group1 packs
// data_size=2B, LDS padding (64B interval, 32B pad => 96B pitch), tensor
// dims/strides and tile dims. Groups 2/3 describe the (degenerate) higher
// dims. This toolchain exposes the 6-arg builtin:
//   (u32x4 g0, i32x8 g1, i32x4 g2, i32x4 g3, i32x8 extra, i32 cpol)
// EXEC is ignored by TDM; call from one wave only, completion is tracked
// with TENSORcnt.
// ---------------------------------------------------------------------------
__device__ inline void tdm_load_tile(const bf16_t* gbase, int row0, int k0,
                                     int nrows, int K, unsigned lds_off) {
  unsigned long long ga =
      (unsigned long long)(const void*)(gbase + (size_t)row0 * K + k0);
  u32x4 g0;
  g0[0] = 1u;                                            // count=1 user D#
  g0[1] = lds_off;                                       // lds_addr (bytes)
  g0[2] = (unsigned)(ga & 0xFFFFFFFFu);                  // global_addr lo
  g0[3] = (unsigned)((ga >> 32) & 0x1FFFFFFu) | (2u << 30);  // hi | type=2

  i32x8 g1;
  g1[0] = (1 << 16)        // data_size = 2 bytes
        | (1 << 20)        // pad_enable
        | (3 << 22)        // pad_interval: 16 DWORDs = 64B (one tile row)
        | (7 << 25);       // pad_amount: 8 DWORDs = 32B
  g1[1] = (K & 0xFFFF) << 16;                            // tensor_dim0 lo16
  g1[2] = ((K >> 16) & 0xFFFF) | ((nrows & 0xFFFF) << 16);  // dim0 hi|dim1 lo
  g1[3] = ((nrows >> 16) & 0xFFFF) | (32 << 16);         // dim1 hi|tile_dim0
  g1[4] = 128;                                           // tile_dim1 | tile_dim2=0
  g1[5] = K;                                             // tensor_dim0_stride lo
  g1[6] = 0;                                             // stride hi | dim1_stride lo
  g1[7] = 0;                                             // dim1_stride hi
  i32x4 g2 = {1, 1, 0, 0};  // tensor_dim2=1, tensor_dim3=1, others unused
  i32x4 g3 = {0, 0, 0, 0};
  i32x8 g4 = {0, 0, 0, 0, 0, 0, 0, 0};  // extra descriptor words: unused
  __builtin_amdgcn_tensor_load_to_lds(g0, g1, g2, g3, g4, 0);
}
#endif

// ---------------------------------------------------------------------------
// f32 -> bf16 conversion (weights)
// ---------------------------------------------------------------------------
__global__ void f2bf_kernel(const float* __restrict__ in, bf16_t* __restrict__ out,
                            long long n) {
  long long i = (long long)blockIdx.x * blockDim.x + threadIdx.x;
  long long stride = (long long)gridDim.x * blockDim.x;
  for (; i < n; i += stride) out[i] = (bf16_t)in[i];
}

// ---------------------------------------------------------------------------
// RMSNorm over D=2048, one token per 256-thread block, bf16 output
// ---------------------------------------------------------------------------
__global__ __launch_bounds__(256) void rmsnorm_kernel(
    const float* __restrict__ x, const float* __restrict__ w,
    bf16_t* __restrict__ out) {
  const int t = blockIdx.x;
  const float* xr = x + (size_t)t * D_MODEL;
  float vals[8];
  float ss = 0.f;
#pragma unroll
  for (int i = 0; i < 8; ++i) {
    float v = xr[threadIdx.x + i * 256];
    vals[i] = v; ss += v * v;
  }
#pragma unroll
  for (int m = 1; m < 32; m <<= 1) ss += __shfl_xor(ss, m, 32);
  __shared__ float red[8];
  if ((threadIdx.x & 31) == 0) red[threadIdx.x >> 5] = ss;
  __syncthreads();
  float tot = 0.f;
#pragma unroll
  for (int i = 0; i < 8; ++i) tot += red[i];
  const float inv = rsqrtf(tot / (float)D_MODEL + RMS_EPS);
#pragma unroll
  for (int i = 0; i < 8; ++i) {
    int c = threadIdx.x + i * 256;
    out[(size_t)t * D_MODEL + c] = (bf16_t)(vals[i] * inv * w[c]);
  }
}

// ---------------------------------------------------------------------------
// Per-head RMSNorm + RoPE, in place on bf16 [B*S, D]. One (token, head) per
// 128-thread block.
// ---------------------------------------------------------------------------
__global__ __launch_bounds__(128) void qknorm_rope_kernel(
    bf16_t* __restrict__ qk, const float* __restrict__ nw,
    const int* __restrict__ pos) {
  const int blk = blockIdx.x;
  const int h = blk % NHEAD;
  const int t = blk / NHEAD;
  bf16_t* row = qk + (size_t)t * D_MODEL + h * HDIM;
  const int i = threadIdx.x;
  float v = (float)row[i];
  float ss = v * v;
#pragma unroll
  for (int m = 1; m < 32; m <<= 1) ss += __shfl_xor(ss, m, 32);
  __shared__ float red[4];
  __shared__ float xs[HDIM];
  if ((i & 31) == 0) red[i >> 5] = ss;
  __syncthreads();
  const float tot = red[0] + red[1] + red[2] + red[3];
  const float xn = v * rsqrtf(tot / (float)HDIM + RMS_EPS) * nw[i];
  xs[i] = xn;
  __syncthreads();
  const float p = (float)pos[t];
  const int j = i & 63;
  const float freq = p * expf(-(float)j * 0.14391156508f);  // ln(10000)/64
  float sv, cv;
  __sincosf(freq, &sv, &cv);
  float o;
  if (i < 64) o = xn * cv - xs[i + 64] * sv;
  else        o = xs[i - 64] * sv + xn * cv;
  row[i] = (bf16_t)o;
}

// ---------------------------------------------------------------------------
// V transpose: [B*S, D] token-major -> [B*H, HD, S] hd-major (so P*V
// B-fragments become contiguous key runs).
// ---------------------------------------------------------------------------
__global__ __launch_bounds__(256) void vtrans_kernel(
    const bf16_t* __restrict__ v, bf16_t* __restrict__ vt) {
  __shared__ bf16_t tile[32][33];
  const int bh = blockIdx.z;
  const int h = bh % NHEAD, b = bh / NHEAD;
  const int s0 = blockIdx.x * 32, d0 = blockIdx.y * 32;
  const int tx = threadIdx.x, ty = threadIdx.y;
  for (int i = ty; i < 32; i += 8)
    tile[i][tx] = v[(size_t)(b * S_LEN + s0 + i) * D_MODEL + h * HDIM + d0 + tx];
  __syncthreads();
  for (int i = ty; i < 32; i += 8)
    vt[((size_t)bh * HDIM + d0 + i) * S_LEN + s0 + tx] = tile[tx][i];
}

// ---------------------------------------------------------------------------
// WMMA GEMM: C[M,N] = A[M,K] @ W[N,K]^T  (+ epilogue)
//   EPI 0: store bf16;  EPI 1: add f32 residual R, store f32;  EPI 2: GELU->bf16
// Block = 256 threads (8 waves), tile 128x128, K-step 32.
// Tiles are staged into dynamic LDS by the Tensor Data Mover (wave 0 issues
// the DMA, double-buffered; s_wait_tensorcnt + barrier hand the tile to the
// other waves), with a cooperative-load fallback if the builtin is absent.
// ---------------------------------------------------------------------------
template <int EPI>
__global__ __launch_bounds__(256) void gemm_wmma_kernel(
    const bf16_t* __restrict__ A, const bf16_t* __restrict__ W,
    const float* __restrict__ R, float* __restrict__ outF,
    bf16_t* __restrict__ outB, int M, int N, int K) {
  extern __shared__ __align__(128) bf16_t smem[];  // 2 x (A tile | W tile)
  const int tid  = threadIdx.x;
  const int lane = tid & 31, wave = tid >> 5;
  const int wm = wave >> 2, wn = wave & 3;      // 2 x 4 wave grid
  const int half = lane >> 4, lr = lane & 15;
  const int m0 = blockIdx.y * 128;
  const int n0 = blockIdx.x * 128;

  const v8f vzero = {};
  v8f acc[4][2];
#pragma unroll
  for (int a = 0; a < 4; ++a)
#pragma unroll
    for (int bn = 0; bn < 2; ++bn) acc[a][bn] = vzero;

#if USE_TDM
  if (wave == 0) {
    tdm_load_tile(A, m0, 0, M, K, 0u);
    tdm_load_tile(W, n0, 0, N, K, (unsigned)TILE_BYTES);
    __builtin_amdgcn_s_wait_tensorcnt(0);
  }
  __syncthreads();
#endif

  int cur = 0;
  for (int k0 = 0; k0 < K; k0 += 32) {
    bf16_t* At = smem + (size_t)cur * 2 * TILE_ELEMS;
    bf16_t* Wt = At + TILE_ELEMS;

#if USE_TDM
    if (wave == 0 && k0 + 32 < K) {  // DMA next tiles into the other buffer
      const unsigned nb = (unsigned)((cur ^ 1) * 2 * TILE_BYTES);
      tdm_load_tile(A, m0, k0 + 32, M, K, nb);
      tdm_load_tile(W, n0, k0 + 32, N, K, nb + TILE_BYTES);
    }
#else
    __syncthreads();
#pragma unroll
    for (int it = 0; it < 2; ++it) {
      int i = tid + it * 256;
      int r = i >> 2, c = (i & 3) * 8;
      *(v8bf*)&At[r * TILE_PITCH + c] = *(const v8bf*)&A[(size_t)(m0 + r) * K + k0 + c];
      *(v8bf*)&Wt[r * TILE_PITCH + c] = *(const v8bf*)&W[(size_t)(n0 + r) * K + k0 + c];
    }
    if (k0 + 32 < K) {  // global_prefetch_b8 for next K tile
      __builtin_prefetch(&A[(size_t)(m0 + (tid >> 1)) * K + k0 + 32 + (tid & 1) * 16], 0, 0);
      __builtin_prefetch(&W[(size_t)(n0 + (tid >> 1)) * K + k0 + 32 + (tid & 1) * 16], 0, 0);
    }
    __syncthreads();
#endif

    v16bf af[4], bfm[2];
#pragma unroll
    for (int mf = 0; mf < 4; ++mf)
      af[mf] = load_a_frag(&At[(wm * 64 + mf * 16 + lr) * TILE_PITCH + half * 16]);
#pragma unroll
    for (int nf = 0; nf < 2; ++nf)
      bfm[nf] = *(const v16bf*)&Wt[(wn * 32 + nf * 16 + lr) * TILE_PITCH + half * 16];

#pragma unroll
    for (int mf = 0; mf < 4; ++mf)
#pragma unroll
      for (int nf = 0; nf < 2; ++nf)
        acc[mf][nf] = __builtin_amdgcn_wmma_f32_16x16x32_bf16(
            false, af[mf], false, bfm[nf], (short)0, acc[mf][nf], false, false);

#if USE_TDM
    if (wave == 0) __builtin_amdgcn_s_wait_tensorcnt(0);
    __syncthreads();
    cur ^= 1;
#endif
  }

  // C layout: VGPR r -> row r (lanes 0-15) / r+8 (lanes 16-31), col = lane%16
#pragma unroll
  for (int mf = 0; mf < 4; ++mf)
#pragma unroll
    for (int nf = 0; nf < 2; ++nf)
#pragma unroll
      for (int r = 0; r < 8; ++r) {
        const int row = m0 + wm * 64 + mf * 16 + r + half * 8;
        const int col = n0 + wn * 32 + nf * 16 + lr;
        float val = acc[mf][nf][r];
        if (EPI == 1) {
          val += R[(size_t)row * N + col];
          outF[(size_t)row * N + col] = val;
        } else if (EPI == 2) {
          val = 0.5f * val * (1.0f + erff(val * 0.70710678118f));
          outB[(size_t)row * N + col] = (bf16_t)val;
        } else {
          outB[(size_t)row * N + col] = (bf16_t)val;
        }
      }
}

// ---------------------------------------------------------------------------
// Fused ReLU^2 attention. Each wave owns 16 queries of one (b,h) and streams
// over all keys: scores via WMMA, ReLU^2 in C-fragment, weights re-staged
// through a per-wave LDS tile into A-layout for the P*V WMMA. Denominator is
// a plain running sum (no max trick needed) reduced by shfl_xor at the end.
// ---------------------------------------------------------------------------
__global__ __launch_bounds__(128) void attn_kernel(
    const bf16_t* __restrict__ qb, const bf16_t* __restrict__ kb,
    const bf16_t* __restrict__ vt, bf16_t* __restrict__ attn) {
  const int lane = threadIdx.x & 31;
  const int wave = threadIdx.x >> 5;
  const int task = blockIdx.x * 4 + wave;
  const int qt = task % (S_LEN / 16);
  const int bh = task / (S_LEN / 16);
  const int h = bh % NHEAD;
  const int b = bh / NHEAD;
  const int half = lane >> 4, lr = lane & 15;

  __shared__ __align__(32) bf16_t stage[4][16][48];
  bf16_t(*st)[48] = stage[wave];

  // Q fragments: 4 K-steps of 32 head-dims
  const bf16_t* qrow = qb + (size_t)(b * S_LEN + qt * 16 + lr) * D_MODEL + h * HDIM;
  v16bf qa[4];
#pragma unroll
  for (int ks = 0; ks < 4; ++ks)
    qa[ks] = load_a_frag(qrow + ks * 32 + half * 16);

  const v8f vzero = {};
  v8f num[8];
#pragma unroll
  for (int nf = 0; nf < 8; ++nf) num[nf] = vzero;
  float den[8] = {0.f, 0.f, 0.f, 0.f, 0.f, 0.f, 0.f, 0.f};

  const bf16_t* kbase = kb + (size_t)(b * S_LEN) * D_MODEL + h * HDIM;
  const bf16_t* vbase = vt + (size_t)bh * HDIM * S_LEN;
  const float scale = 0.08838834764831845f;  // 1/sqrt(128)

  for (int blk = 0; blk < S_LEN / 32; ++blk) {
#pragma unroll
    for (int kt = 0; kt < 2; ++kt) {
      const bf16_t* krow = kbase + (size_t)(blk * 32 + kt * 16 + lr) * D_MODEL;
      v8f s = vzero;
#pragma unroll
      for (int ks = 0; ks < 4; ++ks) {
        v16bf bfrag = *(const v16bf*)(krow + ks * 32 + half * 16);
        s = __builtin_amdgcn_wmma_f32_16x16x32_bf16(false, qa[ks], false, bfrag,
                                                    (short)0, s, false, false);
      }
#pragma unroll
      for (int r = 0; r < 8; ++r) {
        float w = s[r] * scale;
        w = (w > 0.f) ? w * w : 0.f;
        den[r] += w;
        st[r + half * 8][kt * 16 + lr] = (bf16_t)w;  // C-layout -> tile
      }
    }
    // intra-wave LDS RAW: DS ops are in-order per wave; fence the compiler
    // and drain the DS counter before re-reading in A-layout.
    asm volatile("s_wait_dscnt 0" ::: "memory");
    v16bf wa = load_a_frag(&st[lr][half * 16]);

#pragma unroll
    for (int nf = 0; nf < 8; ++nf) {
      v16bf vfrag = *(const v16bf*)(vbase + (size_t)(nf * 16 + lr) * S_LEN +
                                    blk * 32 + half * 16);
      num[nf] = __builtin_amdgcn_wmma_f32_16x16x32_bf16(false, wa, false, vfrag,
                                                        (short)0, num[nf], false, false);
    }
  }

  // reduce denominator across the 16 lanes sharing each query row
  float rden[8];
#pragma unroll
  for (int r = 0; r < 8; ++r) {
    float d = den[r];
#pragma unroll
    for (int m = 1; m < 16; m <<= 1) d += __shfl_xor(d, m, 32);
    rden[r] = 1.0f / fmaxf(d, ATTN_EPS);
  }

  bf16_t* orow = attn + (size_t)(b * S_LEN + qt * 16) * D_MODEL + h * HDIM;
#pragma unroll
  for (int nf = 0; nf < 8; ++nf)
#pragma unroll
    for (int r = 0; r < 8; ++r)
      orow[(size_t)(r + half * 8) * D_MODEL + nf * 16 + lr] =
          (bf16_t)(num[nf][r] * rden[r]);
}

// ---------------------------------------------------------------------------
// Host orchestration
// ---------------------------------------------------------------------------
extern "C" void kernel_launch(void* const* d_in, const int* in_sizes, int n_in,
                              void* d_out, int out_size, void* d_ws, size_t ws_size,
                              hipStream_t stream) {
  const float* x   = (const float*)d_in[0];
  const int*   pos = (const int*)d_in[1];
  const float* n1w = (const float*)d_in[2];
  const float* n2w = (const float*)d_in[3];
  const float* qnw = (const float*)d_in[4];
  const float* knw = (const float*)d_in[5];
  const float* qw  = (const float*)d_in[6];
  const float* kw  = (const float*)d_in[7];
  const float* vw  = (const float*)d_in[8];
  const float* ow  = (const float*)d_in[9];
  const float* w1  = (const float*)d_in[10];
  const float* w2  = (const float*)d_in[11];
  float* out = (float*)d_out;

  const int M  = in_sizes[0] / D_MODEL;  // B*S
  const int Bb = M / S_LEN;

  const size_t DD = (size_t)D_MODEL * D_MODEL;
  const size_t DF = (size_t)D_MODEL * FF_DIM;
  const size_t MD = (size_t)M * D_MODEL;
  const size_t MF = (size_t)M * FF_DIM;

  char* pw = (char*)d_ws;
  auto alloc = [&](size_t elems) -> bf16_t* {
    bf16_t* r = (bf16_t*)pw;
    pw += (elems * sizeof(bf16_t) + 255) & ~(size_t)255;
    return r;
  };
  bf16_t* wq_b   = alloc(DD);
  bf16_t* wk_b   = alloc(DD);
  bf16_t* wv_b   = alloc(DD);
  bf16_t* wo_b   = alloc(DD);
  bf16_t* w1_b   = alloc(DF);
  bf16_t* w2_b   = alloc(DF);
  bf16_t* h_b    = alloc(MD);
  bf16_t* q_b    = alloc(MD);
  bf16_t* k_b    = alloc(MD);
  bf16_t* v_b    = alloc(MD);
  bf16_t* vt_b   = alloc(MD);
  bf16_t* attn_b = alloc(MD);
  bf16_t* h2_b   = alloc(MD);
  bf16_t* g_b    = alloc(MF);
  if ((size_t)(pw - (char*)d_ws) > ws_size) return;  // insufficient scratch

  // 1) weights -> bf16
  auto cvt = [&](const float* s, bf16_t* d, size_t n) {
    f2bf_kernel<<<dim3(4096), dim3(256), 0, stream>>>(s, d, (long long)n);
  };
  cvt(qw, wq_b, DD); cvt(kw, wk_b, DD); cvt(vw, wv_b, DD); cvt(ow, wo_b, DD);
  cvt(w1, w1_b, DF); cvt(w2, w2_b, DF);

  // 2) pre-attention norm
  rmsnorm_kernel<<<dim3(M), dim3(256), 0, stream>>>(x, n1w, h_b);

  // 3) Q/K/V projections (bf16 out)
  dim3 gqkv(D_MODEL / 128, M / 128);
  gemm_wmma_kernel<0><<<gqkv, dim3(256), GEMM_LDS_BYTES, stream>>>(h_b, wq_b, nullptr, nullptr, q_b, M, D_MODEL, D_MODEL);
  gemm_wmma_kernel<0><<<gqkv, dim3(256), GEMM_LDS_BYTES, stream>>>(h_b, wk_b, nullptr, nullptr, k_b, M, D_MODEL, D_MODEL);
  gemm_wmma_kernel<0><<<gqkv, dim3(256), GEMM_LDS_BYTES, stream>>>(h_b, wv_b, nullptr, nullptr, v_b, M, D_MODEL, D_MODEL);

  // 4) per-head RMSNorm + RoPE (in place)
  qknorm_rope_kernel<<<dim3(M * NHEAD), dim3(128), 0, stream>>>(q_b, qnw, pos);
  qknorm_rope_kernel<<<dim3(M * NHEAD), dim3(128), 0, stream>>>(k_b, knw, pos);

  // 5) V -> [B*H, HD, S]
  vtrans_kernel<<<dim3(S_LEN / 32, HDIM / 32, Bb * NHEAD), dim3(32, 8), 0, stream>>>(v_b, vt_b);

  // 6) fused ReLU^2 attention
  attn_kernel<<<dim3((Bb * NHEAD * (S_LEN / 16)) / 4), dim3(128), 0, stream>>>(q_b, k_b, vt_b, attn_b);

  // 7) output projection + residual (writes x2 into d_out, f32)
  gemm_wmma_kernel<1><<<gqkv, dim3(256), GEMM_LDS_BYTES, stream>>>(attn_b, wo_b, x, out, nullptr, M, D_MODEL, D_MODEL);

  // 8) pre-FFN norm
  rmsnorm_kernel<<<dim3(M), dim3(256), 0, stream>>>(out, n2w, h2_b);

  // 9) FFN up-proj + exact GELU (bf16 out)
  gemm_wmma_kernel<2><<<dim3(FF_DIM / 128, M / 128), dim3(256), GEMM_LDS_BYTES, stream>>>(h2_b, w1_b, nullptr, nullptr, g_b, M, FF_DIM, D_MODEL);

  // 10) FFN down-proj + residual (reads/writes d_out element-wise)
  gemm_wmma_kernel<1><<<dim3(D_MODEL / 128, M / 128), dim3(256), GEMM_LDS_BYTES, stream>>>(g_b, w2_b, out, out, nullptr, M, D_MODEL, FF_DIM);
}
